// CrossAttentionWithPositionalEncoding_48498770707174
// MI455X (gfx1250) — matrix operand order repaired and checked by reading, not compiled, not run on hardware
//
#include <hip/hip_runtime.h>
#include <hip/hip_bf16.h>
#include <math.h>

typedef _Float16 half_t;
typedef __attribute__((ext_vector_type(16))) _Float16 v16h;
typedef __attribute__((ext_vector_type(8)))  float    v8f;

#define D_MODEL  256
#define N_BATCH  4
#define NP_TOK   1024
#define HW_TOK   4096
#define N_HEADS  8
#define DH_HEAD  32
#define LN10000  9.2103403719761836f

// ---------------------------------------------------------------------------
// WMMA fragment loaders, wave32; A and B share contiguous k = 16*(lane/16) + s.
// ---------------------------------------------------------------------------

// A-matrix 16x32 f16 (MxK), row-major source with stride ld (halves).
// lanes 0-15: M=lane, K=0..15 ; lanes 16-31: M=lane-16, K=16..31

__device__ __forceinline__ v16h load_a_frag(const half_t* src, int ld, int lane) {
  int m  = lane & 15;
  int kb = (lane >> 4) << 4;            // 0 or 16
  const half_t* p = src + (size_t)m * ld;
  v16h f;
#pragma unroll
  for (int i = 0; i < 8; ++i) f[i]     = p[kb + i];
#pragma unroll
  for (int i = 0; i < 8; ++i) f[8 + i] = p[kb + 8 + i];
  return f;
}

// B-matrix 32x16 f16 (KxN). Memory holds B transposed: src[n*ld + k] == B[k][n].
// lanes 0-15: N=lane, K=0..15 ; lanes 16-31: N=lane-16, K=16..31
__device__ __forceinline__ v16h load_b_frag(const half_t* src, int ld, int lane) {
  int n  = lane & 15;
  int kb = (lane >> 4) << 4;            // 0 or 16
  const half_t* p = src + (size_t)n * ld + kb;
  v16h f;
#pragma unroll
  for (int i = 0; i < 16; ++i) f[i] = p[i];
  return f;
}

__device__ __forceinline__ v8f wmma_f16(v16h a, v16h b, v8f c) {
  return __builtin_amdgcn_wmma_f32_16x16x32_f16(false, a, false, b, (short)0, c,
                                                false, false);
}

// ---------------------------------------------------------------------------
// f32 -> f16 weight pre-conversion (so GEMM inner loops carry no v_cvt work)
// ---------------------------------------------------------------------------
__global__ void __launch_bounds__(256)
k_cvt_f16(const float* __restrict__ src, half_t* __restrict__ dst) {
  int idx  = blockIdx.x * 256 + threadIdx.x;
  dst[idx] = (half_t)src[idx];
}

// ---------------------------------------------------------------------------
// Positional encoding: pe[p, 2m] = sin(x*div_m), pe[p, 2m+1] = cos(y*div_m),
// div_m = exp(-2m*ln(1e4)/D), x=col/(G-1), y=row/(G-1), p = row*G + col.
// ---------------------------------------------------------------------------
__device__ __forceinline__ float pos_enc(int pos, int j, int gshift, float invLast) {
  int grid = 1 << gshift;
  int row  = pos >> gshift;
  int col  = pos & (grid - 1);
  int m    = j >> 1;
  float dv = __expf((float)m * (-2.0f * LN10000 / (float)D_MODEL));
  if (j & 1) return __cosf(((float)row * invLast) * dv);
  return __sinf(((float)col * invLast) * dv);
}

// lf_pe (f32 for residual, f16 for GEMM) = local_feat + posenc over 32x32 grid
__global__ void __launch_bounds__(256)
k_pe_local(const float* __restrict__ lf, float* __restrict__ out32,
           half_t* __restrict__ out16) {
  int idx = blockIdx.x * 256 + threadIdx.x;          // over N*NP*D
  int j   = idx & (D_MODEL - 1);
  int p   = (idx >> 8) & (NP_TOK - 1);
  float v = lf[idx] + pos_enc(p, j, 5, 1.0f / 31.0f);
  out32[idx] = v;
  out16[idx] = (half_t)v;
}

// gf_pe f16 = transpose(global_feat) + posenc over 64x64 grid
__global__ void __launch_bounds__(256)
k_pe_global(const float* __restrict__ gf, half_t* __restrict__ out16) {
  int idx = blockIdx.x * 256 + threadIdx.x;          // over N*HW*D, d fastest
  int d   = idx & (D_MODEL - 1);
  int hw  = (idx >> 8) & (HW_TOK - 1);
  int b   = idx >> 20;                               // HW*D = 2^20
  float v = gf[((size_t)(b * D_MODEL + d)) * HW_TOK + hw]
            + pos_enc(hw, d, 6, 1.0f / 63.0f);
  out16[idx] = (half_t)v;
}

// ---------------------------------------------------------------------------
// GEMM: out[row, col] = sum_k A_f16[row,k] * Wh_f16[col,k] + bias[col]
// A: (M x 256) f16 row-major.  Wh: (256 x 256) f16 row-major (used as W^T),
// so Wh serves directly as the transposed-B source for load_b_frag.
// Block = 128 threads (4 waves), tile 64x64; wave w -> rows [r0, r0+16).
// transOut: store f16 as out[col*M + row] (for V^T); else row-major.
// ---------------------------------------------------------------------------
__global__ void __launch_bounds__(128)
k_gemm(const half_t* __restrict__ A, const half_t* __restrict__ Wh,
       const float* __restrict__ bias, half_t* __restrict__ outH,
       float* __restrict__ outF, int M, int transOut) {
  int lane = threadIdx.x & 31;
  int wave = threadIdx.x >> 5;
  int r0   = blockIdx.x * 64 + wave * 16;
  int c0   = blockIdx.y * 64;

  v8f acc[4] = {};
#pragma unroll
  for (int k0 = 0; k0 < D_MODEL; k0 += 32) {
    v16h a = load_a_frag(A + (size_t)r0 * D_MODEL + k0, D_MODEL, lane);
#pragma unroll
    for (int nt = 0; nt < 4; ++nt) {
      v16h b = load_b_frag(Wh + (size_t)(c0 + nt * 16) * D_MODEL + k0,
                           D_MODEL, lane);
      acc[nt] = wmma_f16(a, b, acc[nt]);
    }
  }

  int nlo = lane & 15;
  int mhi = (lane >> 4) << 3;
#pragma unroll
  for (int nt = 0; nt < 4; ++nt) {
    int col  = c0 + nt * 16 + nlo;
    float bv = bias[col];
#pragma unroll
    for (int v = 0; v < 8; ++v) {
      int row   = r0 + mhi + v;
      float val = acc[nt][v] + bv;
      if (outH) {
        if (transOut) outH[(size_t)col * M + row] = (half_t)val;
        else          outH[(size_t)row * D_MODEL + col] = (half_t)val;
      }
      if (outF) outF[(size_t)row * D_MODEL + col] = val;
    }
  }
}

// ---------------------------------------------------------------------------
// Flash attention: 1 wave handles 16 queries for one (head, batch).
// Per 32-key chunk: 2 score WMMAs (K=DH=32), online softmax with half-wave
// shuffle reductions, probs relaid C->A through LDS, 2 PV WMMAs (Vt layout
// makes B-fragments contiguous). Epilogue: o/l + residual -> f16 for O-proj.
// ---------------------------------------------------------------------------
__global__ void __launch_bounds__(32)
k_attn(const half_t* __restrict__ Q, const half_t* __restrict__ K,
       const half_t* __restrict__ Vt, const float* __restrict__ lf32,
       half_t* __restrict__ X2) {
  __shared__ half_t Pl[16 * 32];

  const int lane = threadIdx.x;
  const int qt   = blockIdx.x;
  const int head = blockIdx.y;
  const int b    = blockIdx.z;
  const float scale = 0.17677669529663687f;          // 1/sqrt(32)

  const half_t* qp = Q + ((size_t)(b * NP_TOK + qt * 16)) * D_MODEL + head * DH_HEAD;
  v16h aq = load_a_frag(qp, D_MODEL, lane);

  v8f o0 = {}, o1 = {};
  float mrow[8], lrow[8];
#pragma unroll
  for (int v = 0; v < 8; ++v) { mrow[v] = -INFINITY; lrow[v] = 0.0f; }

  const half_t* kb_ = K + ((size_t)b * HW_TOK) * D_MODEL + head * DH_HEAD;
  const size_t vld  = (size_t)N_BATCH * HW_TOK;      // Vt row stride
  const half_t* vb0 = Vt + (size_t)(head * DH_HEAD) * vld + (size_t)b * HW_TOK;
  const half_t* vb1 = vb0 + (size_t)16 * vld;

  const int row = ((lane >> 4) << 3);
  const int col = lane & 15;

  for (int kc = 0; kc < HW_TOK; kc += 32) {
    v16h bk0 = load_b_frag(kb_ + (size_t)kc * D_MODEL, D_MODEL, lane);
    v16h bk1 = load_b_frag(kb_ + (size_t)(kc + 16) * D_MODEL, D_MODEL, lane);
    v8f z0 = {}, z1 = {};
    v8f s0 = wmma_f16(aq, bk0, z0);
    v8f s1 = wmma_f16(aq, bk1, z1);

#pragma unroll
    for (int v = 0; v < 8; ++v) {
      float x0 = s0[v] * scale;
      float x1 = s1[v] * scale;
      float mx = fmaxf(x0, x1);
#pragma unroll
      for (int msk = 8; msk >= 1; msk >>= 1)
        mx = fmaxf(mx, __shfl_xor(mx, msk, 32));     // stays in half-wave
      float mnew = fmaxf(mrow[v], mx);
      float p0 = __expf(x0 - mnew);
      float p1 = __expf(x1 - mnew);
      float sum = p0 + p1;
#pragma unroll
      for (int msk = 8; msk >= 1; msk >>= 1)
        sum += __shfl_xor(sum, msk, 32);
      float sf = __expf(mrow[v] - mnew);             // 0 on first chunk
      lrow[v] = lrow[v] * sf + sum;
      mrow[v] = mnew;
      o0[v] *= sf;
      o1[v] *= sf;
      // stage probs (C layout -> LDS row-major 16x32); DS ops are in-order
      Pl[(row + v) * 32 + col]      = (half_t)p0;
      Pl[(row + v) * 32 + col + 16] = (half_t)p1;
    }

    v16h ap  = load_a_frag(Pl, 32, lane);
    v16h bv0 = load_b_frag(vb0 + kc, (int)vld, lane);
    v16h bv1 = load_b_frag(vb1 + kc, (int)vld, lane);
    o0 = wmma_f16(ap, bv0, o0);
    o1 = wmma_f16(ap, bv1, o1);
  }

#pragma unroll
  for (int v = 0; v < 8; ++v) {
    int q = qt * 16 + row + v;
    float inv = 1.0f / lrow[v];
    size_t base = ((size_t)(b * NP_TOK + q)) * D_MODEL + head * DH_HEAD;
    X2[base + col]      = (half_t)(o0[v] * inv + lf32[base + col]);
    X2[base + 16 + col] = (half_t)(o1[v] * inv + lf32[base + 16 + col]);
  }
}

// ---------------------------------------------------------------------------
// Host-side orchestration
// ---------------------------------------------------------------------------
extern "C" void kernel_launch(void* const* d_in, const int* in_sizes, int n_in,
                              void* d_out, int out_size, void* d_ws, size_t ws_size,
                              hipStream_t stream) {
  const float* lf = (const float*)d_in[0];
  const float* gf = (const float*)d_in[1];
  const float* Wq = (const float*)d_in[2];
  const float* bq = (const float*)d_in[3];
  const float* Wk = (const float*)d_in[4];
  const float* bk = (const float*)d_in[5];
  const float* Wv = (const float*)d_in[6];
  const float* bv = (const float*)d_in[7];
  const float* Wo = (const float*)d_in[8];
  const float* bo = (const float*)d_in[9];
  float* out = (float*)d_out;

  char* ws = (char*)d_ws;
  const size_t nLoc = (size_t)N_BATCH * NP_TOK * D_MODEL;   // 1,048,576
  const size_t nGlb = (size_t)N_BATCH * HW_TOK * D_MODEL;   // 4,194,304
  const size_t nW   = (size_t)D_MODEL * D_MODEL;            // 65,536

  size_t off = 0;
  float*  lf_pe32 = (float*)(ws + off);  off += nLoc * sizeof(float);
  half_t* lf_pe16 = (half_t*)(ws + off); off += nLoc * sizeof(half_t);
  half_t* gf_pe16 = (half_t*)(ws + off); off += nGlb * sizeof(half_t);
  half_t* Qf      = (half_t*)(ws + off); off += nLoc * sizeof(half_t);
  half_t* Kf      = (half_t*)(ws + off); off += nGlb * sizeof(half_t);
  half_t* Vtf     = (half_t*)(ws + off); off += nGlb * sizeof(half_t);
  half_t* X2      = (half_t*)(ws + off); off += nLoc * sizeof(half_t);
  half_t* Wqh     = (half_t*)(ws + off); off += nW * sizeof(half_t);
  half_t* Wkh     = (half_t*)(ws + off); off += nW * sizeof(half_t);
  half_t* Wvh     = (half_t*)(ws + off); off += nW * sizeof(half_t);
  half_t* Woh     = (half_t*)(ws + off); off += nW * sizeof(half_t);
  (void)off; (void)ws_size; (void)n_in; (void)in_sizes; (void)out_size;

  // 0) pre-convert weights to f16 once (keeps GEMM inner loops cvt-free)
  k_cvt_f16<<<dim3((unsigned)(nW / 256)), dim3(256), 0, stream>>>(Wq, Wqh);
  k_cvt_f16<<<dim3((unsigned)(nW / 256)), dim3(256), 0, stream>>>(Wk, Wkh);
  k_cvt_f16<<<dim3((unsigned)(nW / 256)), dim3(256), 0, stream>>>(Wv, Wvh);
  k_cvt_f16<<<dim3((unsigned)(nW / 256)), dim3(256), 0, stream>>>(Wo, Woh);

  // 1) positional encodings (+transpose for global features)
  k_pe_local<<<dim3((unsigned)(nLoc / 256)), dim3(256), 0, stream>>>(lf, lf_pe32, lf_pe16);
  k_pe_global<<<dim3((unsigned)(nGlb / 256)), dim3(256), 0, stream>>>(gf, gf_pe16);

  // 2) projections (WMMA GEMMs); V stored transposed for attention B-frags
  const int Mq = N_BATCH * NP_TOK;    // 4096
  const int Mk = N_BATCH * HW_TOK;    // 16384
  k_gemm<<<dim3(Mq / 64, D_MODEL / 64), dim3(128), 0, stream>>>(
      lf_pe16, Wqh, bq, Qf, nullptr, Mq, 0);
  k_gemm<<<dim3(Mk / 64, D_MODEL / 64), dim3(128), 0, stream>>>(
      gf_pe16, Wkh, bk, Kf, nullptr, Mk, 0);
  k_gemm<<<dim3(Mk / 64, D_MODEL / 64), dim3(128), 0, stream>>>(
      gf_pe16, Wvh, bv, Vtf, nullptr, Mk, 1);

  // 3) flash attention + residual -> X2 (f16)
  k_attn<<<dim3(NP_TOK / 16, N_HEADS, N_BATCH), dim3(32), 0, stream>>>(
      Qf, Kf, Vtf, lf_pe32, X2);

  // 4) output projection -> f32 result
  k_gemm<<<dim3(Mq / 64, D_MODEL / 64), dim3(128), 0, stream>>>(
      X2, Woh, bo, nullptr, out, Mq, 0);
}
